// KBNet_22883585753704
// MI455X (gfx1250) — matrix operand order repaired and checked by reading, not compiled, not run on hardware
//
#include <hip/hip_runtime.h>
#include <hip/hip_bf16.h>
#include <math.h>

// ---------------------------------------------------------------------------
// CDNA5 / gfx1250 implementation of the KBNet relational-GAT reference.
// All dense projections use v_wmma_f32_16x16x32_f16 (wave32 WMMA).
// Block = 512 threads = 16 waves; each wave owns <=2 output 16-col tiles so
// the WMMA body is fully unrolled with compile-time bounds (no spills).
// ---------------------------------------------------------------------------

typedef __attribute__((ext_vector_type(16))) _Float16 half16;
typedef __attribute__((ext_vector_type(8)))  float    float8;

#define NEG_SLOPE 0.01f
#define EPS_NRM   1e-12f

// ------------------------------ utilities ----------------------------------
__global__ void k_zero(float* p, size_t n) {
  size_t i = (size_t)blockIdx.x * blockDim.x + threadIdx.x;
  size_t stride = (size_t)gridDim.x * blockDim.x;
  for (; i < n; i += stride) p[i] = 0.f;
}

// One wave32 per row: out = row / max(||row||, eps)
__global__ void k_l2norm_rows(const float* __restrict__ in, float* __restrict__ out,
                              int rows, int dim) {
  int gw   = (int)(((size_t)blockIdx.x * blockDim.x + threadIdx.x) >> 5);
  int lane = threadIdx.x & 31;
  if (gw >= rows) return;
  const float* r = in + (size_t)gw * dim;
  float ssq = 0.f;
  for (int j = lane; j < dim; j += 32) { float v = r[j]; ssq += v * v; }
#pragma unroll
  for (int m = 16; m >= 1; m >>= 1) ssq += __shfl_xor(ssq, m, 32);
  float s = 1.f / fmaxf(sqrtf(ssq), EPS_NRM);
  float* o = out + (size_t)gw * dim;
  for (int j = lane; j < dim; j += 32) o[j] = r[j] * s;
}

// One wave32 per row: v = leaky_relu(v); out = v / max(||v||, eps)  (in-place safe)
__global__ void k_lrelu_l2norm_rows(const float* __restrict__ in, float* __restrict__ out,
                                    int rows, int dim) {
  int gw   = (int)(((size_t)blockIdx.x * blockDim.x + threadIdx.x) >> 5);
  int lane = threadIdx.x & 31;
  if (gw >= rows) return;
  const float* r = in + (size_t)gw * dim;
  float ssq = 0.f;
  for (int j = lane; j < dim; j += 32) {
    float v = r[j]; v = (v > 0.f) ? v : NEG_SLOPE * v; ssq += v * v;
  }
#pragma unroll
  for (int m = 16; m >= 1; m >>= 1) ssq += __shfl_xor(ssq, m, 32);
  float s = 1.f / fmaxf(sqrtf(ssq), EPS_NRM);
  float* o = out + (size_t)gw * dim;
  for (int j = lane; j < dim; j += 32) {
    float v = r[j]; v = (v > 0.f) ? v : NEG_SLOPE * v; o[j] = v * s;
  }
}

// Pack W (row-major [out_dim][in_dim], computing feat @ W^T) into the WMMA
// B-fragment layout for v_wmma_f32_16x16x32_f16:
//   B tile (kt,nt): lane = (n%16) | ((k%32)/16)<<4 ; half h = k%16
//   Wpk[((nt*KT + kt)*32 + lane)*16 + h] = W[n][k]  (zero padded)
__global__ void k_pack_w(const float* __restrict__ W, _Float16* __restrict__ Wpk,
                         int in_dim, int out_dim, int KT, int NT) {
  size_t total  = (size_t)NT * KT * 512;
  size_t stride = (size_t)gridDim.x * blockDim.x;
  for (size_t idx = (size_t)blockIdx.x * blockDim.x + threadIdx.x; idx < total; idx += stride) {
    int h    = (int)(idx & 15);
    int lane = (int)((idx >> 4) & 31);
    size_t t = idx >> 9;
    int kt   = (int)(t % KT);
    int nt   = (int)(t / KT);
    int ncol = nt * 16 + (lane & 15);
    int k    = kt * 32 + ((lane >> 4) << 4) + h;
    float v  = (k < in_dim && ncol < out_dim) ? W[(size_t)ncol * in_dim + k] : 0.f;
    Wpk[idx] = (_Float16)v;
  }
}

// ------------------------ WMMA tile body (specialized) ----------------------
// Computes CNT 16x16 output tiles for this wave: nt = wave + q*16.
// A fragment comes from LDS row-major f16 tile via two b128 loads per K-chunk.
template <int KPAD, int CNT>
__device__ __forceinline__ void wmma_body(const _Float16 (*smA)[KPAD],
                                          const _Float16* __restrict__ Wpk,
                                          const float* __restrict__ bias,
                                          int out_dim, int wave, int lane,
                                          float* __restrict__ dst,
                                          int dst_row0, int row_limit) {
  constexpr int KT = KPAD / 32;
  const int mrow = lane & 15;   // C/B column & A row
  const int sel  = lane >> 4;   // lanes 16..31 carry K 16..31 of each 32-K chunk

  float8 acc[CNT];
#pragma unroll
  for (int q = 0; q < CNT; ++q) acc[q] = (float8){0.f,0.f,0.f,0.f,0.f,0.f,0.f,0.f};

  const _Float16* bp[CNT];
#pragma unroll
  for (int q = 0; q < CNT; ++q)
    bp[q] = Wpk + (size_t)(wave + q * 16) * KT * 512 + (size_t)lane * 16;

  for (int kt = 0; kt < KT; ++kt) {
    // A fragment: K=sel*16..sel*16+15, contiguous in LDS
    union { int4 w[2]; half16 h; } ua;
    const int4* rp = (const int4*)&smA[mrow][kt * 32];
    ua.w[0] = rp[2 * sel];
    ua.w[1] = rp[2 * sel + 1];
    half16 a = ua.h;

    if (kt + 1 < KT)
      __builtin_prefetch((const void*)(bp[0] + 512), 0, 1);

#pragma unroll
    for (int q = 0; q < CNT; ++q) {
      half16 b = *(const half16*)bp[q];
      acc[q] = __builtin_amdgcn_wmma_f32_16x16x32_f16(
          false, a, false, b, (short)0, acc[q], false, false);
      bp[q] += 512;
    }
  }

#pragma unroll
  for (int q = 0; q < CNT; ++q) {
    int ncol = (wave + q * 16) * 16 + mrow;
    if (ncol < out_dim) {
      float bb = bias ? bias[ncol] : 0.f;
#pragma unroll
      for (int rr = 0; rr < 8; ++rr) {        // C layout: vgpr rr -> M = rr + sel*8
        int grow = dst_row0 + rr + (sel << 3);
        if (grow < row_limit)
          dst[(size_t)grow * out_dim + ncol] = acc[q][rr] + bb;
      }
    }
  }
}

// Wave-uniform dispatch on how many N-tiles this wave owns (0, 1 or 2).
template <int KPAD>
__device__ __forceinline__ void wmma_tiles(const _Float16 (*smA)[KPAD],
                                           const _Float16* __restrict__ Wpk,
                                           const float* __restrict__ bias,
                                           int out_dim, int n_tiles,
                                           float* __restrict__ dst,
                                           int dst_row0, int row_limit) {
  const int wave = threadIdx.x >> 5;   // 0..15
  const int lane = threadIdx.x & 31;
  const int cnt  = (wave < n_tiles) ? (((n_tiles - 1 - wave) >> 4) + 1) : 0;
  if (cnt >= 2)
    wmma_body<KPAD, 2>(smA, Wpk, bias, out_dim, wave, lane, dst, dst_row0, row_limit);
  else if (cnt == 1)
    wmma_body<KPAD, 1>(smA, Wpk, bias, out_dim, wave, lane, dst, dst_row0, row_limit);
}

// ------------------- per-edge projection: c = fc1([h_i,h_j,g_k]) -----------
template <int KPAD>
__global__ void __launch_bounds__(512)
k_edge_gemm(const float* __restrict__ hsrc, int dh,
            const float* __restrict__ gmat, int dg,
            const int* __restrict__ rowi, const int* __restrict__ coli,
            const int* __restrict__ etype, int E_,
            const _Float16* __restrict__ Wpk,
            const float* __restrict__ bias,
            int out_dim, int n_tiles,
            float* __restrict__ cout) {
  __shared__ _Float16 smA[16][KPAD];
  const int tile = blockIdx.x;
  const int tid  = threadIdx.x;
  const int r  = tid >> 5;   // 0..15 : edge row within tile
  const int j0 = tid & 31;

  int e = tile * 16 + r;
  int ri = 0, ci = 0, ti = 0;
  bool valid = (e < E_);
  if (valid) { ri = rowi[e]; ci = coli[e]; ti = etype[e]; }
  for (int j = j0; j < KPAD; j += 32) {
    float v = 0.f;
    if (valid) {
      if (j < dh)               v = hsrc[(size_t)ri * dh + j];
      else if (j < 2 * dh)      v = hsrc[(size_t)ci * dh + (j - dh)];
      else if (j < 2 * dh + dg) v = gmat[(size_t)ti * dg + (j - 2 * dh)];
    }
    smA[r][j] = (_Float16)v;
  }
  __syncthreads();
  wmma_tiles<KPAD>(smA, Wpk, bias, out_dim, n_tiles, cout, tile * 16, E_);
}

// ------------------- dense rows GEMM: out = src @ W^T + b ------------------
template <int KPAD>
__global__ void __launch_bounds__(512)
k_dense_gemm(const float* __restrict__ src, int dim,
             const _Float16* __restrict__ Wpk,
             const float* __restrict__ bias,
             int out_dim, int n_tiles,
             float* __restrict__ outp, int M) {
  __shared__ _Float16 smA[16][KPAD];
  const int tile = blockIdx.x;
  const int tid  = threadIdx.x;
  const int r  = tid >> 5;
  const int j0 = tid & 31;
  int m = tile * 16 + r;
  for (int j = j0; j < KPAD; j += 32) {
    float v = 0.f;
    if (m < M && j < dim) v = src[(size_t)m * dim + j];
    smA[r][j] = (_Float16)v;
  }
  __syncthreads();
  wmma_tiles<KPAD>(smA, Wpk, bias, out_dim, n_tiles, outp, tile * 16, M);
}

// -------- attention pass 1: ev = exp(leaky_relu(att . c)); rowsum += ev ----
__global__ void k_att_ev(const float* __restrict__ c, const float* __restrict__ att,
                         const int* __restrict__ rowi,
                         float* __restrict__ ev, float* __restrict__ rs,
                         int E_, int H, int D) {
  int gw   = (int)(((size_t)blockIdx.x * blockDim.x + threadIdx.x) >> 5);
  int lane = threadIdx.x & 31;
  if (gw >= E_ * H) return;
  int e = gw / H, h = gw % H;
  const float* ce = c + (size_t)e * H * D + (size_t)h * D;
  float a = 0.f;
  for (int d = lane; d < D; d += 32) a += ce[d] * att[h * D + d];
#pragma unroll
  for (int m = 16; m >= 1; m >>= 1) a += __shfl_xor(a, m, 32);
  if (lane == 0) {
    float bv  = (a > 0.f) ? a : NEG_SLOPE * a;
    float evv = expf(bv);
    ev[(size_t)e * H + h] = evv;
    atomicAdd(&rs[(size_t)rowi[e] * H + h], evv);
  }
}

// -------- attention pass 2: hout[col] += (ev/rowsum[row]) * c --------------
__global__ void k_aggregate(const float* __restrict__ c, const float* __restrict__ ev,
                            const float* __restrict__ rs,
                            const int* __restrict__ rowi, const int* __restrict__ coli,
                            float* __restrict__ hout, int E_, int H, int D) {
  int e = blockIdx.x;
  if (e >= E_) return;
  int re = rowi[e], ce = coli[e];
  int od = H * D;
  for (int j = threadIdx.x; j < od; j += blockDim.x) {
    int h = j / D;
    float alpha = ev[(size_t)e * H + h] / rs[(size_t)re * H + h];
    atomicAdd(&hout[(size_t)ce * od + j], alpha * c[(size_t)e * od + j]);
  }
}

// -------- final merge: l2norm(ent + h2n per head) then mean over heads -----
__global__ void k_final(const float* __restrict__ h2n,   // [N, H*D] (already lrelu+l2norm)
                        const float* __restrict__ ent,   // [N, D]
                        float* __restrict__ out,         // [N, D]
                        int n, int H, int D) {
  __shared__ float red[2][128];
  __shared__ float obuf[200];
  int node = blockIdx.x;
  if (node >= n) return;
  int h = threadIdx.x >> 7;      // 0 or 1 (H == 2)
  int t = threadIdx.x & 127;

  float ssq = 0.f;
  for (int d = t; d < D; d += 128) {
    float v = ent[(size_t)node * D + d] + h2n[(size_t)node * H * D + (size_t)h * D + d];
    ssq += v * v;
  }
  red[h][t] = ssq;
  __syncthreads();
  for (int s = 64; s > 0; s >>= 1) {
    if (t < s) red[h][t] += red[h][t + s];
    __syncthreads();
  }
  float scale = (1.f / fmaxf(sqrtf(red[h][0]), EPS_NRM)) / (float)H;

  if (h == 0)
    for (int d = t; d < D; d += 128) {
      float v = ent[(size_t)node * D + d] + h2n[(size_t)node * H * D + d];
      obuf[d] = v * scale;
    }
  __syncthreads();
  if (h == 1)
    for (int d = t; d < D; d += 128) {
      float v = ent[(size_t)node * D + d] + h2n[(size_t)node * H * D + (size_t)D + d];
      obuf[d] += v * scale;
    }
  __syncthreads();
  for (int d = threadIdx.x; d < D; d += 256)
    out[(size_t)node * D + d] = obuf[d];
}

// ---------------------------------------------------------------------------
extern "C" void kernel_launch(void* const* d_in, const int* in_sizes, int n_in,
                              void* d_out, int out_size, void* d_ws, size_t ws_size,
                              hipStream_t stream) {
  const float* x    = (const float*)d_in[0];
  const float* g    = (const float*)d_in[1];
  const int*   eidx = (const int*)d_in[2];
  const int*   etyp = (const int*)d_in[3];
  const float* W1   = (const float*)d_in[4];
  const float* b1   = (const float*)d_in[5];
  const float* att1 = (const float*)d_in[6];
  const float* W2   = (const float*)d_in[7];
  const float* b2   = (const float*)d_in[8];
  const float* att2 = (const float*)d_in[9];
  const float* We   = (const float*)d_in[10];
  const float* be   = (const float*)d_in[11];
  const float* Wr   = (const float*)d_in[12];
  const float* br   = (const float*)d_in[13];

  const int XS = 100, GS = 100, HID = 100, OUT = 200, H = 2;
  const int N_ = in_sizes[0] / XS;      // 50000
  const int E_ = in_sizes[3];           // 300000
  const int R_ = in_sizes[1] / GS;      // 237
  const int* rowi = eidx;
  const int* coli = eidx + E_;

  // layer dims
  const int IN1 = 2 * XS + GS, OD1 = H * HID;           // 300 -> 200
  const int IN2 = 2 * (H * HID) + GS, OD2 = H * OUT;    // 500 -> 400
  const int KP1 = 320, KT1 = KP1 / 32, NT1 = (OD1 + 15) / 16;   // 10, 13
  const int KP2 = 512, KT2 = KP2 / 32, NT2 = (OD2 + 15) / 16;   // 16, 25
  const int KPE = 128, KTE = KPE / 32, NTE = (OUT + 15) / 16;   // 4, 13

  // ---- workspace carve-out ----
  char* w = (char*)d_ws;
  auto take = [&](size_t bytes) -> void* {
    void* p = (void*)w;
    w += (bytes + 255) & ~(size_t)255;
    return p;
  };
  float*    xn   = (float*)take((size_t)N_ * XS * 4);
  float*    h1   = (float*)take((size_t)N_ * OD1 * 4);
  float*    h2   = (float*)take((size_t)N_ * OD2 * 4);
  float*    ent  = (float*)take((size_t)N_ * OUT * 4);
  float*    ev   = (float*)take((size_t)E_ * H * 4);
  float*    rs   = (float*)take((size_t)N_ * H * 4);
  _Float16* wpk1 = (_Float16*)take((size_t)NT1 * KT1 * 512 * 2);
  _Float16* wpk2 = (_Float16*)take((size_t)NT2 * KT2 * 512 * 2);
  _Float16* wpkE = (_Float16*)take((size_t)NTE * KTE * 512 * 2);
  _Float16* wpkR = (_Float16*)take((size_t)NTE * KTE * 512 * 2);
  float*    cbuf = (float*)take((size_t)E_ * OD2 * 4);   // reused by both layers

  float* out_h = (float*)d_out;                     // [N, OUT]
  float* out_g = (float*)d_out + (size_t)N_ * OUT;  // [R, OUT]

  // ---- pack weights into WMMA B-fragment layout ----
  {
    int thr = 256;
    k_pack_w<<<(int)(((size_t)NT1 * KT1 * 512 + thr - 1) / thr), thr, 0, stream>>>(W1, wpk1, IN1, OD1, KT1, NT1);
    k_pack_w<<<(int)(((size_t)NT2 * KT2 * 512 + thr - 1) / thr), thr, 0, stream>>>(W2, wpk2, IN2, OD2, KT2, NT2);
    k_pack_w<<<(int)(((size_t)NTE * KTE * 512 + thr - 1) / thr), thr, 0, stream>>>(We, wpkE, XS, OUT, KTE, NTE);
    k_pack_w<<<(int)(((size_t)NTE * KTE * 512 + thr - 1) / thr), thr, 0, stream>>>(Wr, wpkR, GS, OUT, KTE, NTE);
  }

  // ---- x normalization ----
  k_l2norm_rows<<<(N_ + 7) / 8, 256, 0, stream>>>(x, xn, N_, XS);

  // ================= layer 1 =================
  k_edge_gemm<KP1><<<(E_ + 15) / 16, 512, 0, stream>>>(
      xn, XS, g, GS, rowi, coli, etyp, E_, wpk1, b1, OD1, NT1, cbuf);

  k_zero<<<2048, 256, 0, stream>>>(rs, (size_t)N_ * H);
  k_att_ev<<<(E_ * H + 7) / 8, 256, 0, stream>>>(cbuf, att1, rowi, ev, rs, E_, H, HID);

  k_zero<<<2048, 256, 0, stream>>>(h1, (size_t)N_ * OD1);
  k_aggregate<<<E_, 256, 0, stream>>>(cbuf, ev, rs, rowi, coli, h1, E_, H, HID);

  k_lrelu_l2norm_rows<<<(N_ * H + 7) / 8, 256, 0, stream>>>(h1, h1, N_ * H, HID);

  // ================= layer 2 =================
  k_edge_gemm<KP2><<<(E_ + 15) / 16, 512, 0, stream>>>(
      h1, H * HID, g, GS, rowi, coli, etyp, E_, wpk2, b2, OD2, NT2, cbuf);

  k_zero<<<2048, 256, 0, stream>>>(rs, (size_t)N_ * H);
  k_att_ev<<<(E_ * H + 7) / 8, 256, 0, stream>>>(cbuf, att2, rowi, ev, rs, E_, H, OUT);

  k_zero<<<2048, 256, 0, stream>>>(h2, (size_t)N_ * OD2);
  k_aggregate<<<E_, 256, 0, stream>>>(cbuf, ev, rs, rowi, coli, h2, E_, H, OUT);

  k_lrelu_l2norm_rows<<<(N_ * H + 7) / 8, 256, 0, stream>>>(h2, h2, N_ * H, OUT);

  // ============ entity / relation linears (WMMA) ============
  k_dense_gemm<KPE><<<(N_ + 15) / 16, 512, 0, stream>>>(xn, XS, wpkE, be, OUT, NTE, ent, N_);
  k_dense_gemm<KPE><<<(R_ + 15) / 16, 512, 0, stream>>>(g, GS, wpkR, br, OUT, NTE, out_g, R_);

  // ============ final merge ============
  k_final<<<N_, 256, 0, stream>>>(h2, ent, out_h, N_, H, OUT);
}
